// MMD_loss_parallel_56667798504272
// MI455X (gfx1250) — matrix operand order repaired and checked by reading, not compiled, not run on hardware
//
#include <hip/hip_runtime.h>
#include <hip/hip_fp16.h>

// MMD loss, MI455X / gfx1250, round 5.
// - Symmetric gram (ty<=tx tiles only), loss = M00^2 + M11^2 - 2*M01^2.
// - Single-exp geometric bandwidth ladder: e + e^2 + e^4 + e^8 + e^16.
// - f16 pre-converted X; GEMM streams b128 f16 tiles (L2-resident).
// - Ping-pong LDS (1 barrier/chunk); scheduling fences pin the pipeline:
//   global loads (k+1) | fence | WMMA block (k) | fence | LDS stores (k+1).
// - k-loop force-unrolled: acc stays SSA-chained (no phi mov storm),
//   LDS offsets fold to immediates, fences keep the staging.

typedef __attribute__((ext_vector_type(16))) _Float16 v16h;
typedef __attribute__((ext_vector_type(8)))  _Float16 h8;
typedef __attribute__((ext_vector_type(8)))  float    v8f;

#define NROWS   8192
#define BATCHSZ 4096
#define DF      512
#define CLAMP_V 20.0f
#define LDS_STRIDE 40          // halves per LDS row (80 B)
#define BUFH (128 * LDS_STRIDE)

union fragu { v16h v; h8 q[2]; };

__device__ __forceinline__ float clampv(float x) {
    return fminf(fmaxf(x, -CLAMP_V), CLAMP_V);
}

__device__ __forceinline__ const float* row_ptr(const float* s0, const float* s1, int r) {
    return (r < BATCHSZ) ? (s0 + (size_t)r * DF) : (s1 + (size_t)(r - BATCHSZ) * DF);
}

// ---- one-shot clip + f32->f16 convert of concat(X) into ws ---------------------
__global__ void convert_kernel(const float* __restrict__ s0, const float* __restrict__ s1,
                               _Float16* __restrict__ Xh) {
    size_t idx = ((size_t)blockIdx.x * 256 + threadIdx.x) * 8;   // 2048 blocks
    int row = (int)(idx >> 9);
    int col = (int)(idx & 511);
    const float* p = row_ptr(s0, s1, row) + col;
    float4 v0 = *(const float4*)p;
    float4 v1 = *(const float4*)(p + 4);
    h8 o = { (_Float16)clampv(v0.x), (_Float16)clampv(v0.y),
             (_Float16)clampv(v0.z), (_Float16)clampv(v0.w),
             (_Float16)clampv(v1.x), (_Float16)clampv(v1.y),
             (_Float16)clampv(v1.z), (_Float16)clampv(v1.w) };
    *(h8*)(Xh + idx) = o;
}

// ---- per-row squared norm from the f16 values (so D_ii ~ 0 consistently) -------
__global__ void rowsq_kernel(const _Float16* __restrict__ Xh, float* __restrict__ sq) {
    int row = blockIdx.x;
    const _Float16* p = Xh + (size_t)row * DF;
    int t = threadIdx.x;                       // 256 threads, 512 cols
    float a = (float)p[t];
    float b = (float)p[t + 256];
    __shared__ float red[256];
    red[t] = a * a + b * b;
    __syncthreads();
    for (int s = 128; s > 0; s >>= 1) {
        if (t < s) red[t] += red[t + s];
        __syncthreads();
    }
    if (t == 0) sq[row] = red[0];
}

// ---- partial column sums (256 row-groups of 32 rows) ---------------------------
__global__ void colpart_kernel(const _Float16* __restrict__ Xh, float* __restrict__ colpart) {
    int g = blockIdx.x;                        // 256 blocks
    int t = threadIdx.x;                       // 256 threads
    int r0 = g * 32;
    for (int half = 0; half < 2; ++half) {
        int c = t + half * 256;
        float sum = 0.f;
        for (int r = 0; r < 32; ++r)
            sum += (float)Xh[(size_t)(r0 + r) * DF + c];
        colpart[g * DF + c] = sum;
    }
}

// ---- bandwidth: sum(D) = 2*N*S1 - 2*sum_d(colsum_d^2) --------------------------
__global__ void bw_kernel(const float* __restrict__ colpart, const float* __restrict__ sq,
                          float* __restrict__ invbw) {
    int t = threadIdx.x;                       // 512 threads
    __shared__ float red[512];
    float cs = 0.f;
    for (int g = 0; g < 256; ++g) cs += colpart[g * DF + t];
    red[t] = cs * cs;
    __syncthreads();
    for (int s = 256; s > 0; s >>= 1) {
        if (t < s) red[t] += red[t + s];
        __syncthreads();
    }
    float sumColSq = red[0];
    __syncthreads();
    float s1 = 0.f;
    for (int j = 0; j < 16; ++j) s1 += sq[t + 512 * j];
    red[t] = s1;
    __syncthreads();
    for (int s = 256; s > 0; s >>= 1) {
        if (t < s) red[t] += red[t + s];
        __syncthreads();
    }
    if (t == 0) {
        float S1 = red[0];
        float n = (float)NROWS;
        float sumD = 2.f * n * S1 - 2.f * sumColSq;
        float bw = (sumD / (n * n - n)) * 0.25f;   // / KERNEL_MUL^(KERNEL_NUM/2)
        float m = 1.f;
        for (int i = 0; i < 5; ++i) { invbw[i] = 1.f / (bw * m); m *= 2.f; }
    }
}

// ---- GEMM helper: 1 A-fragment load + 8 B-fragment loads + 8 WMMAs -------------
__device__ __forceinline__ void mma_step(const _Float16* __restrict__ Abuf,
                                         const _Float16* __restrict__ Bbuf,
                                         int wave, int lo, int hi, v8f acc[8]) {
    // A fragment (16x32 f16): two contiguous 16B chunks per lane.
    fragu af;
    const _Float16* Abase = Abuf + (wave * 16 + lo) * LDS_STRIDE;
    af.q[0] = *(const h8*)(Abase + 8 * hi);
    af.q[1] = *(const h8*)(Abase + 16 + 8 * hi);

    #pragma unroll
    for (int nt = 0; nt < 8; ++nt) {
        // B fragment (32x16 f16): two contiguous 16B chunks per lane.
        fragu bf;
        const _Float16* Bbase = Bbuf + (nt * 16 + lo) * LDS_STRIDE + 8 * hi;
        bf.q[0] = *(const h8*)(Bbase);
        bf.q[1] = *(const h8*)(Bbase + 16);
        acc[nt] = __builtin_amdgcn_wmma_f32_16x16x32_f16(
            false, af.v, false, bf.v, (short)0, acc[nt], false, false);
    }
}

// ---- main WMMA GEMM + exp epilogue, upper-triangle tiles only ------------------
__global__ void __launch_bounds__(256)
mmd_gemm_kernel(const _Float16* __restrict__ Xh,
                const float* __restrict__ sq, const float* __restrict__ invbw,
                float* __restrict__ tilesum) {
    int tx = blockIdx.x;                       // col tile (0..63)
    int ty = blockIdx.y;                       // row tile (0..63)
    if (ty > tx) return;                       // symmetry: K(i,j) == K(j,i)

    __shared__ _Float16 Ah[2 * BUFH];          // ping-pong buffers
    __shared__ _Float16 Bh[2 * BUFH];
    __shared__ float red[256];

    int rowBase = ty * 128, colBase = tx * 128;
    int t    = threadIdx.x;
    int wave = t >> 5;                         // 0..7 -> rows wave*16..+15 of tile
    int lane = t & 31;
    int lo   = lane & 15, hi = lane >> 4;

    const _Float16* Ag = Xh + (size_t)rowBase * DF;
    const _Float16* Bg = Xh + (size_t)colBase * DF;
    int lr = t >> 1;                           // loader row   0..127
    int lc = (t & 1) * 16;                     // loader half-col: 0 or 16
    size_t goff = (size_t)lr * DF + lc;        // + k0 per chunk
    int soff = lr * LDS_STRIDE + lc;           // LDS store offset (halves)

    v8f acc[8];
    for (int i = 0; i < 8; ++i)
        for (int j = 0; j < 8; ++j) acc[i][j] = 0.f;

    // prologue: chunk 0 -> buffer 0
    {
        h8 a0 = *(const h8*)(Ag + goff),     a1 = *(const h8*)(Ag + goff + 8);
        h8 b0 = *(const h8*)(Bg + goff),     b1 = *(const h8*)(Bg + goff + 8);
        *(h8*)(Ah + soff) = a0;  *(h8*)(Ah + soff + 8) = a1;
        *(h8*)(Bh + soff) = b0;  *(h8*)(Bh + soff + 8) = b1;
    }
    __syncthreads();

    // steady state: issue loads (k+1), fence, WMMA (k), fence, store (k+1)
    #pragma unroll
    for (int k = 0; k < (DF / 32) - 1; ++k) {
        int cur = (k & 1) * BUFH;
        int nxt = BUFH - cur;
        size_t g = goff + (size_t)(k + 1) * 32;
        h8 a0 = *(const h8*)(Ag + g),        a1 = *(const h8*)(Ag + g + 8);
        h8 b0 = *(const h8*)(Bg + g),        b1 = *(const h8*)(Bg + g + 8);
        asm volatile("" ::: "memory");         // keep loads issued before WMMA block

        mma_step(Ah + cur, Bh + cur, wave, lo, hi, acc);

        asm volatile("" ::: "memory");         // keep stores after WMMA block
        *(h8*)(Ah + nxt + soff) = a0;  *(h8*)(Ah + nxt + soff + 8) = a1;
        *(h8*)(Bh + nxt + soff) = b0;  *(h8*)(Bh + nxt + soff + 8) = b1;
        __syncthreads();
    }
    mma_step(Ah + ((DF / 32 - 1) & 1) * BUFH, Bh + ((DF / 32 - 1) & 1) * BUFH,
             wave, lo, hi, acc);

    // Epilogue: D = sq_r + sq_c - 2*G, clamp; geometric ladder:
    // e = exp(-D/(bw*16)); sum_i exp(-D/(bw*2^i)) = e + e^2 + e^4 + e^8 + e^16.
    float sqr[8];
    #pragma unroll
    for (int v = 0; v < 8; ++v)
        sqr[v] = sq[rowBase + wave * 16 + 8 * hi + v];
    float ib = invbw[4];                       // 1/(bw*16)

    float lsum = 0.f;
    #pragma unroll
    for (int nt = 0; nt < 8; ++nt) {
        float sqc = sq[colBase + nt * 16 + lo];
        #pragma unroll
        for (int v = 0; v < 8; ++v) {
            float d = fmaxf(sqr[v] + sqc - 2.f * acc[nt][v], 0.f);
            float e  = __expf(-d * ib);
            float e2 = e * e, e4 = e2 * e2, e8 = e4 * e4, e16 = e8 * e8;
            lsum += e + e2 + e4 + e8 + e16;
        }
    }

    red[t] = lsum;
    __syncthreads();
    for (int s = 128; s > 0; s >>= 1) {
        if (t < s) red[t] += red[t + s];
        __syncthreads();
    }
    if (t == 0) tilesum[ty * 64 + tx] = red[0] * 0.2f;   // / KERNEL_NUM
}

// ---- deterministic weighted reduction: 0:(0,0) 1:(0,1) 2:(1,1) -----------------
__global__ void blocksum_kernel(const float* __restrict__ tilesum, float* __restrict__ S) {
    int ab = blockIdx.x;                       // 3 blocks
    int t = threadIdx.x;                       // 256 threads
    float s = 0.f;
    for (int j = 0; j < 4; ++j) {
        int l = t + j * 256;                   // 0..1023
        int lty = l >> 5, ltx = l & 31;
        if (ab == 1) {                         // off-diag block: all tiles ty<tx, weight 1
            s += tilesum[lty * 64 + (32 + ltx)];
        } else {                               // diag blocks: ty<tx weight 2, ty==tx weight 1
            int off = (ab == 2) ? 32 : 0;
            if (lty < ltx)       s += 2.f * tilesum[(off + lty) * 64 + (off + ltx)];
            else if (lty == ltx) s +=       tilesum[(off + lty) * 64 + (off + ltx)];
        }
    }
    __shared__ float red[256];
    red[t] = s;
    __syncthreads();
    for (int st = 128; st > 0; st >>= 1) {
        if (t < st) red[t] += red[t + st];
        __syncthreads();
    }
    if (t == 0) S[ab] = red[0] / ((float)BATCHSZ * (float)BATCHSZ);
}

// ---- final scalar: loss = M00^2 + M11^2 - 2*M01^2 ------------------------------
__global__ void loss_kernel(const float* __restrict__ S, float* __restrict__ out) {
    if (threadIdx.x == 0 && blockIdx.x == 0) {
        float M00 = S[0], M01 = S[1], M11 = S[2];
        out[0] = M00 * M00 + M11 * M11 - 2.f * M01 * M01;
    }
}

extern "C" void kernel_launch(void* const* d_in, const int* in_sizes, int n_in,
                              void* d_out, int out_size, void* d_ws, size_t ws_size,
                              hipStream_t stream) {
    (void)in_sizes; (void)n_in; (void)out_size; (void)ws_size;
    const float* s0 = (const float*)d_in[0];
    const float* s1 = (const float*)d_in[1];

    // ws layout: Xh[8192*512] f16 (8MB, 16B-aligned first) |
    //            sq[8192] | colpart[256*512] | invbw[8] | tilesum[4096] | S[4]
    _Float16* Xh   = (_Float16*)d_ws;
    float* fws     = (float*)d_ws + (size_t)NROWS * DF / 2;
    float* sq      = fws;
    float* colpart = sq + NROWS;
    float* invbw   = colpart + 256 * DF;
    float* tilesum = invbw + 8;
    float* S       = tilesum + 4096;

    convert_kernel <<<2048, 256, 0, stream>>>(s0, s1, Xh);
    rowsq_kernel   <<<NROWS, 256, 0, stream>>>(Xh, sq);
    colpart_kernel <<<256,   256, 0, stream>>>(Xh, colpart);
    bw_kernel      <<<1,     512, 0, stream>>>(colpart, sq, invbw);
    mmd_gemm_kernel<<<dim3(64, 64), 256, 0, stream>>>(Xh, sq, invbw, tilesum);
    blocksum_kernel<<<3,     256, 0, stream>>>(tilesum, S);
    loss_kernel    <<<1,      64, 0, stream>>>(S, (float*)d_out);
}
